// EnhancedRecurrentGCN_78941498901099
// MI455X (gfx1250) — matrix operand order repaired and checked by reading, not compiled, not run on hardware
//
#include <hip/hip_runtime.h>

// ---------------------------------------------------------------------------
// Fused DCRNN (K=1 -> pure GEMM) inference for MI455X / gfx1250, wave32 WMMA.
//   cell(x) with h0=0:  (1 - sigmoid(x@Wz+bz)) * tanh(x@Wh+bh)   (r-gate dead)
//   out = relu(relu(cell2(relu(cell1(x)))) @ Wl1 + bl1) @ wl2 + bl2
// Weights: W[0,0]+W[1,0], first Fin rows only (h0 = 0 kills the rest).
//
// Round-2 changes: B-fragments come from LDS (double-buffered layer-1 weight
// slices staged cooperatively per K-step; layer-2/head weights staged once),
// A-stream (x) software-pipelined one K-step ahead, head reduction via
// __shfl_xor (no LDS round-trip). LDS ~131 KB -> 2 blocks/WGP co-residency.
// ---------------------------------------------------------------------------

typedef __attribute__((ext_vector_type(16))) _Float16 v16h;
typedef __attribute__((ext_vector_type(8)))  _Float16 v8h;
typedef __attribute__((ext_vector_type(8)))  float    v8f;
typedef __attribute__((ext_vector_type(4)))  float    v4f;

#define N_ROWS 100000
#define F_IN   256
#define HID    128
#define HID2   64

// workspace layout (offsets in _Float16 elements) -- N-major (row = out col)
#define OFF_W1Z 0        // [128][256]
#define OFF_W1H 32768    // [128][256]
#define OFF_W2Z 65536    // [64][128]
#define OFF_W2H 73728    // [64][128]
#define OFF_WL1 81920    // [16][64]
#define PREP_TOTAL 82944

// LDS strides (halfs); padded so B-fragment lanes land on distinct banks
#define LDS1 40    // layer-1 slice row: 32 valid + 8 pad
#define LDS2 136   // layer-2 row: 128 valid + 8 pad
#define LDSL 72    // head row: 64 valid + 8 pad
#define LD1  136   // H1 activations
#define LD2  72    // H2 activations
#define SLICE_HALFS (2 * 128 * LDS1)   // z+h slice, 10240 halfs = 20 KB

__device__ __forceinline__ v8f wmma_f16(v16h a, v16h b, v8f c) {
    return __builtin_amdgcn_wmma_f32_16x16x32_f16(
        false, a, false, b, (short)0, c, false, false);
}

// A-fragment (16x32 f16) from f32 global rows.
// Lane l<16 : row m0+l, K = k0..k0+7 , k0+16..k0+23
// Lane l+16 : row m0+l, K = k0+8..15, k0+24..31
__device__ __forceinline__ v16h load_a_global_f32(const float* xrow, int k0, int hi) {
    const float* p = xrow + k0 + hi * 8;
    v4f a0 = *(const v4f*)(p);
    v4f a1 = *(const v4f*)(p + 4);
    v4f a2 = *(const v4f*)(p + 16);
    v4f a3 = *(const v4f*)(p + 20);
    v16h r;
#pragma unroll
    for (int i = 0; i < 4; ++i) {
        r[i]      = (_Float16)a0[i];
        r[4 + i]  = (_Float16)a1[i];
        r[8 + i]  = (_Float16)a2[i];
        r[12 + i] = (_Float16)a3[i];
    }
    return r;
}

// A-fragment from row-major f16 LDS (ld multiple of 8 halfs -> 16B rows)
__device__ __forceinline__ v16h load_a_lds(const _Float16* base, int ld, int row,
                                           int k0, int hi) {
    const _Float16* p = base + row * ld + k0 + hi * 8;
    v8h lo = *(const v8h*)(p);
    v8h hh = *(const v8h*)(p + 16);
    v16h r;
#pragma unroll
    for (int i = 0; i < 8; ++i) { r[i] = lo[i]; r[8 + i] = hh[i]; }
    return r;
}

// B-fragment (32x16 f16) from N-major weights (global or LDS), ldk = K stride.
// Lane l<16 : col n0+l, K = k0..k0+7 , k0+16..k0+23 ;  lane l+16 : col n0+l, K = k0+8..15, k0+24..31
__device__ __forceinline__ v16h load_b(const _Float16* wT, int ldk, int n0,
                                       int k0, int lane) {
    const _Float16* p = wT + (n0 + (lane & 15)) * ldk + k0 + ((lane >> 4) << 3);
    v8h lo = *(const v8h*)(p);
    v8h hh = *(const v8h*)(p + 16);
    v16h r;
#pragma unroll
    for (int i = 0; i < 8; ++i) { r[i] = lo[i]; r[8 + i] = hh[i]; }
    return r;
}

__device__ __forceinline__ float sigmoid_f(float x) {
    return 0.5f + 0.5f * tanhf(0.5f * x);   // one v_tanh_f32
}

// --------------------------- weight prep -----------------------------------
__global__ void prep_weights(const float* __restrict__ wz1,
                             const float* __restrict__ wh1,
                             const float* __restrict__ wz2,
                             const float* __restrict__ wh2,
                             const float* __restrict__ wl1,
                             _Float16* __restrict__ ws) {
    int i = blockIdx.x * blockDim.x + threadIdx.x;
    if (i < 32768) {                        // W1zT[n][k], n<128, k<256
        int n = i >> 8, k = i & 255;
        ws[OFF_W1Z + i] = (_Float16)(wz1[k * 128 + n] + wz1[384 * 128 + k * 128 + n]);
    } else if (i < 65536) {                 // W1hT
        int j = i - 32768; int n = j >> 8, k = j & 255;
        ws[OFF_W1H + j] = (_Float16)(wh1[k * 128 + n] + wh1[384 * 128 + k * 128 + n]);
    } else if (i < 73728) {                 // W2zT[n][k], n<64, k<128
        int j = i - 65536; int n = j >> 7, k = j & 127;
        ws[OFF_W2Z + j] = (_Float16)(wz2[k * 64 + n] + wz2[192 * 64 + k * 64 + n]);
    } else if (i < 81920) {                 // W2hT
        int j = i - 73728; int n = j >> 7, k = j & 127;
        ws[OFF_W2H + j] = (_Float16)(wh2[k * 64 + n] + wh2[192 * 64 + k * 64 + n]);
    } else if (i < PREP_TOTAL) {            // Wl1T[n][k], n<16, k<64
        int j = i - 81920; int n = j >> 6, k = j & 63;
        ws[OFF_WL1 + j] = (_Float16)(wl1[k * 16 + n]);
    }
}

// Stage one layer-1 K-slice (cols 0..127, K k0..k0+31, z then h) into LDS.
__device__ __forceinline__ void stage_w1_slice(const _Float16* __restrict__ ws,
                                               _Float16* dst, int k0, int tid) {
#pragma unroll
    for (int c = 0; c < 4; ++c) {
        int idx = tid + (c << 8);            // 0..1023
        int mat = idx >> 9;                  // 0 = z, 1 = h
        int n   = (idx >> 2) & 127;
        int o   = (idx & 3) << 3;
        const _Float16* src = ws + (mat ? OFF_W1H : OFF_W1Z) + n * F_IN + k0 + o;
        *(v8h*)(dst + mat * (128 * LDS1) + n * LDS1 + o) = *(const v8h*)src;
    }
}

// --------------------------- fused forward ---------------------------------
// Block: 256 threads = 8 waves; each wave owns a 16-row strip; 128 rows/block.
__global__ __launch_bounds__(256, 1)
void dcrnn_fused(const float* __restrict__ x, const _Float16* __restrict__ ws,
                 const float* __restrict__ bz1, const float* __restrict__ bh1,
                 const float* __restrict__ bz2, const float* __restrict__ bh2,
                 const float* __restrict__ bl1, const float* __restrict__ wl2,
                 const float* __restrict__ bl2, float* __restrict__ out) {
    __shared__ _Float16 sW1buf[2][SLICE_HALFS];  // 40960 B, double-buffered z+h
    __shared__ _Float16 sW2z[64 * LDS2];         // 17408 B
    __shared__ _Float16 sW2h[64 * LDS2];         // 17408 B
    __shared__ _Float16 sWl1[16 * LDSL];         //  2304 B
    __shared__ _Float16 sh1[128 * LD1];          // 34816 B
    __shared__ _Float16 sh2[128 * LD2];          // 18432 B
                                                 // total 131,328 B

    const int tid  = threadIdx.x;
    const int wave = tid >> 5;
    const int lane = tid & 31;
    const int l16  = lane & 15;
    const int hi   = lane >> 4;
    const int m0   = wave << 4;

    int myRow = blockIdx.x * 128 + m0 + l16;     // clamp for tail tile (loads only)
    if (myRow >= N_ROWS) myRow = N_ROWS - 1;
    const float* xrow = x + (size_t)myRow * F_IN;

    // ---- stage layer-2/head weights (once) + layer-1 slice 0 ----
    for (int c = tid; c < 2176; c += 256) {
        if (c < 1024) {            // W2z: 64 rows x 16 chunks
            int n = c >> 4, o = (c & 15) << 3;
            *(v8h*)(sW2z + n * LDS2 + o) = *(const v8h*)(ws + OFF_W2Z + n * HID + o);
        } else if (c < 2048) {     // W2h
            int j = c - 1024; int n = j >> 4, o = (j & 15) << 3;
            *(v8h*)(sW2h + n * LDS2 + o) = *(const v8h*)(ws + OFF_W2H + n * HID + o);
        } else {                   // Wl1: 16 rows x 8 chunks
            int j = c - 2048; int n = j >> 3, o = (j & 7) << 3;
            *(v8h*)(sWl1 + n * LDSL + o) = *(const v8h*)(ws + OFF_WL1 + n * HID2 + o);
        }
    }
    stage_w1_slice(ws, sW1buf[0], 0, tid);
    __syncthreads();

    // ---- Layer 1: H1 = relu((1-sigmoid(X Wz)) * tanh(X Wh)), 16x128 strip ----
    v8f accz[8], acch[8];
#pragma unroll
    for (int nt = 0; nt < 8; ++nt) { accz[nt] = (v8f)0.0f; acch[nt] = (v8f)0.0f; }

    v16h a_cur = load_a_global_f32(xrow, 0, hi);   // prefetched K-step 0
#pragma unroll 1
    for (int kt = 0; kt < 8; ++kt) {               // K = 256, step 32
        if (kt < 7) stage_w1_slice(ws, sW1buf[(kt + 1) & 1], (kt + 1) << 5, tid);
        const int knext = (kt < 7 ? kt + 1 : 7) << 5;
        v16h a_nxt = load_a_global_f32(xrow, knext, hi);   // prefetch A
        const _Float16* bufz = sW1buf[kt & 1];
        const _Float16* bufh = bufz + 128 * LDS1;
#pragma unroll
        for (int nt = 0; nt < 8; ++nt) {
            v16h bz = load_b(bufz, LDS1, nt << 4, 0, lane);
            v16h bh = load_b(bufh, LDS1, nt << 4, 0, lane);
            accz[nt] = wmma_f16(a_cur, bz, accz[nt]);
            acch[nt] = wmma_f16(a_cur, bh, acch[nt]);
        }
        a_cur = a_nxt;
        __syncthreads();   // slice double-buffer handoff (uniform)
    }
#pragma unroll
    for (int nt = 0; nt < 8; ++nt) {
        const int n = (nt << 4) + l16;
        const float vbz = bz1[n], vbh = bh1[n];
#pragma unroll
        for (int i = 0; i < 8; ++i) {
            float z  = sigmoid_f(accz[nt][i] + vbz);
            float ht = tanhf(acch[nt][i] + vbh);
            float h  = fmaxf((1.0f - z) * ht, 0.0f);
            sh1[(m0 + i + (hi << 3)) * LD1 + n] = (_Float16)h;  // wave-private strip
        }
    }

    // ---- Layer 2: H2 = relu((1-sigmoid(H1 Wz2)) * tanh(H1 Wh2)), 16x64 ----
    v8f acc2z[4], acc2h[4];
#pragma unroll
    for (int nt = 0; nt < 4; ++nt) { acc2z[nt] = (v8f)0.0f; acc2h[nt] = (v8f)0.0f; }
#pragma unroll 1
    for (int kt = 0; kt < 4; ++kt) {               // K = 128, step 32
        const int k0 = kt << 5;
        v16h a = load_a_lds(sh1, LD1, m0 + l16, k0, hi);
#pragma unroll
        for (int nt = 0; nt < 4; ++nt) {
            v16h bz = load_b(sW2z, LDS2, nt << 4, k0, lane);
            v16h bh = load_b(sW2h, LDS2, nt << 4, k0, lane);
            acc2z[nt] = wmma_f16(a, bz, acc2z[nt]);
            acc2h[nt] = wmma_f16(a, bh, acc2h[nt]);
        }
    }
#pragma unroll
    for (int nt = 0; nt < 4; ++nt) {
        const int n = (nt << 4) + l16;
        const float vbz = bz2[n], vbh = bh2[n];
#pragma unroll
        for (int i = 0; i < 8; ++i) {
            float z  = sigmoid_f(acc2z[nt][i] + vbz);
            float ht = tanhf(acc2h[nt][i] + vbh);
            float h  = fmaxf((1.0f - z) * ht, 0.0f);
            sh2[(m0 + i + (hi << 3)) * LD2 + n] = (_Float16)h;
        }
    }

    // ---- Head GEMM: L = relu(H2 @ Wl1 + bl1), 16x16 per strip ----
    v8f acc3 = (v8f)0.0f;
#pragma unroll
    for (int kt = 0; kt < 2; ++kt) {               // K = 64, step 32
        const int k0 = kt << 5;
        v16h a = load_a_lds(sh2, LD2, m0 + l16, k0, hi);
        v16h b = load_b(sWl1, LDSL, 0, k0, lane);
        acc3 = wmma_f16(a, b, acc3);
    }

    // ---- Final 16-wide dot via shfl_xor butterfly (no LDS, no barrier) ----
    {
        const float vb = bl1[l16];
        const float w  = wl2[l16];
        const float b2 = bl2[0];
#pragma unroll
        for (int i = 0; i < 8; ++i) {
            float s = fmaxf(acc3[i] + vb, 0.0f) * w;   // L[m][l16] * wl2[l16]
            s += __shfl_xor(s, 1);
            s += __shfl_xor(s, 2);
            s += __shfl_xor(s, 4);
            s += __shfl_xor(s, 8);                     // sum over 16-lane group
            if (l16 == 0) {
                int row = blockIdx.x * 128 + m0 + (hi << 3) + i;
                if (row < N_ROWS) out[row] = s + b2;
            }
        }
    }
}

// ---------------------------------------------------------------------------
extern "C" void kernel_launch(void* const* d_in, const int* in_sizes, int n_in,
                              void* d_out, int out_size, void* d_ws, size_t ws_size,
                              hipStream_t stream) {
    const float* x    = (const float*)d_in[0];
    // d_in[1]=edge_index, d_in[2]=edge_weight : unused (K=1 DConv, no propagation)
    const float* wz1  = (const float*)d_in[3];
    const float* bz1  = (const float*)d_in[4];
    // d_in[5]/d_in[6] = w_r1/b_r1 : dead (h0 == 0)
    const float* wh1  = (const float*)d_in[7];
    const float* bh1  = (const float*)d_in[8];
    const float* wz2  = (const float*)d_in[9];
    const float* bz2  = (const float*)d_in[10];
    // d_in[11]/d_in[12] = w_r2/b_r2 : dead
    const float* wh2  = (const float*)d_in[13];
    const float* bh2  = (const float*)d_in[14];
    const float* wl1  = (const float*)d_in[15];
    const float* bl1  = (const float*)d_in[16];
    const float* wl2  = (const float*)d_in[17];
    const float* bl2  = (const float*)d_in[18];
    float* out        = (float*)d_out;
    _Float16* ws      = (_Float16*)d_ws;   // needs 165,888 B of workspace

    prep_weights<<<(PREP_TOTAL + 255) / 256, 256, 0, stream>>>(wz1, wh1, wz2, wh2, wl1, ws);

    const int grid = (N_ROWS + 127) / 128;  // 782 blocks x 128 rows
    dcrnn_fused<<<grid, 256, 0, stream>>>(x, ws, bz1, bh1, bz2, bh2, bl1, wl2, bl2, out);
}
